// DeformConv2d_55568286875715
// MI455X (gfx1250) — compile-verified, matrix operand order checked by reading, not run on hardware
//
#include <hip/hip_runtime.h>
#include <math.h>

// ---------------------------------------------------------------------------
// DCNv2 forward for MI455X (gfx1250, wave32, WMMA f32_16x16x32_f16)
//   B=8, Cin=Cout=64, H=W=128, K=3 (KK=9), pad=1
// Pipeline:
//   k0a: x fp32 NCHW -> f16 channel-last [B][H][W][64]   (vector-friendly gathers)
//   k0b: weights fp32 -> f16 row-major [M][K=576], K = kk*64 + c
//   k1 : offset conv (M=32 padded from 27): b128 im2col-in-LDS + WMMA,
//        writes dy/dx/sigmoid(mask) planes [B][9][H][W]
//   k2 : bilinear corner/weight precompute -> packed-f16 modulated gather
//        -> WMMA GEMM (M=64 x N=64 px x K=576) -> out
// ---------------------------------------------------------------------------

typedef _Float16 v8h  __attribute__((ext_vector_type(8)));
typedef _Float16 v16h __attribute__((ext_vector_type(16)));
typedef float    v8f  __attribute__((ext_vector_type(8)));
typedef int      v4i  __attribute__((ext_vector_type(4)));
typedef float    v4f  __attribute__((ext_vector_type(4)));

#define HH     128
#define WW_    128
#define CIN    64
#define COUT   64
#define KDIM   576            // 9 * CIN, ordered K = kk*64 + c
#define KPITCH 584            // LDS K pitch (halves): 16B-aligned rows, de-skewed banks
#define HWSZ   (HH * WW_)

// Build a 16x32 f16 WMMA fragment for this lane from a row-major [.][K] buffer.
// Per ISA layout: halves 0..7 = K (kb + (lane/16)*8 ..), halves 8..15 = +16.
__device__ __forceinline__ v16h make_frag(const _Float16* __restrict__ p0,
                                          const _Float16* __restrict__ p1) {
  v8h lo = *(const v8h*)p0;   // 16B load
  v8h hi = *(const v8h*)p1;   // 16B load
  v16h r;
#pragma unroll
  for (int i = 0; i < 8; ++i) { r[i] = lo[i]; r[8 + i] = hi[i]; }
  return r;
}

__device__ __forceinline__ v8h splat8(float w) {
  const _Float16 h = (_Float16)w;
  v8h r;
#pragma unroll
  for (int i = 0; i < 8; ++i) r[i] = h;
  return r;
}

// ---------------- kernel 0a: x NCHW fp32 -> channel-last f16 ----------------
extern "C" __global__ void __launch_bounds__(256)
xcvt_kernel(const float* __restrict__ x, _Float16* __restrict__ x16) {
  const int idx = blockIdx.x * 256 + threadIdx.x;       // [0, B*H*W)
  const int b = idx >> 14, hw = idx & (HWSZ - 1);
  const float* xb = x + (size_t)b * CIN * HWSZ + hw;    // lane-coalesced reads
  _Float16* dst = x16 + (size_t)idx * CIN;
#pragma unroll
  for (int c8 = 0; c8 < 8; ++c8) {
    v8h v;
#pragma unroll
    for (int j = 0; j < 8; ++j) v[j] = (_Float16)xb[(size_t)(c8 * 8 + j) * HWSZ];
    *(v8h*)(dst + c8 * 8) = v;                          // 16B store
  }
}

// ------------- kernel 0b: weight repack (K permuted to kk*64+c) -------------
extern "C" __global__ void __launch_bounds__(256)
wconvert_kernel(const float* __restrict__ w_off, const float* __restrict__ w_dcn,
                _Float16* __restrict__ Aoff, _Float16* __restrict__ Adcn) {
  const int i = blockIdx.x * 256 + threadIdx.x;
  if (i >= 64 * KDIM) return;
  const int m = i / KDIM, r = i - m * KDIM;
  const int kk = r >> 6, c = r & 63;
  const int src = c * 9 + kk;                           // reference flat K order
  Adcn[i] = (_Float16)w_dcn[m * KDIM + src];
  if (m < 32) Aoff[i] = (m < 27) ? (_Float16)w_off[m * KDIM + src] : (_Float16)0.0f;
}

// ------------------- kernel 1: offset/mask conv (WMMA) ----------------------
extern "C" __global__ void __launch_bounds__(256)
offset_conv_kernel(const _Float16* __restrict__ x16, const float* __restrict__ b_off,
                   const _Float16* __restrict__ Aoff,
                   float* __restrict__ dy, float* __restrict__ dx,
                   float* __restrict__ msk) {
  extern __shared__ _Float16 sB[];        // [64 px][KPITCH] f16 im2col patches
  const int tile = blockIdx.x;            // 256 tiles of 8x8
  const int b    = blockIdx.y;
  const int ty0  = (tile >> 4) * 8, tx0 = (tile & 15) * 8;
  const int t    = threadIdx.x;

  { // --- im2col: pure 16B copies from channel-last x16 (zero pad via select) ---
    const int p  = t & 63, cg = t >> 6;
    const int py = ty0 + (p >> 3), px = tx0 + (p & 7);
    const _Float16* xb = x16 + (size_t)b * HWSZ * CIN + cg * 16;
    const v8h vzero = {};
#pragma unroll
    for (int kk = 0; kk < 9; ++kk) {
      const int iy = py - 1 + kk / 3;
      const int ix = px - 1 + kk % 3;
      const bool ok = (iy >= 0) && (iy < HH) && (ix >= 0) && (ix < WW_);
      const int sp = ok ? (iy * WW_ + ix) : 0;          // safe clamped address
      const _Float16* src = xb + (size_t)sp * CIN;
      v8h a0 = *(const v8h*)src;
      v8h a1 = *(const v8h*)(src + 8);
      a0 = ok ? a0 : vzero;
      a1 = ok ? a1 : vzero;
      _Float16* d = sB + p * KPITCH + kk * 64 + cg * 16;
      *(v8h*)d = a0;
      *(v8h*)(d + 8) = a1;
    }
  }
  __syncthreads();

  // --- GEMM: M=32 (27 live) x N=64 x K=576; 8 waves, one 16x16 tile each ---
  const int wave  = t >> 5, lane = t & 31;
  const int lhalf = lane >> 4, lmod = lane & 15;
  const int mbase = (wave & 1) * 16;
  const int nbase = (wave >> 1) * 16;
  v8f acc = {};
  const _Float16* arow = Aoff + (size_t)(mbase + lmod) * KDIM;
  const _Float16* brow = sB   + (size_t)(nbase + lmod) * KPITCH;
#pragma unroll 2
  for (int kb = 0; kb < KDIM; kb += 32) {
    const v16h a  = make_frag(arow + kb + lhalf * 8, arow + kb + 16 + lhalf * 8);
    const v16h bm = make_frag(brow + kb + lhalf * 8, brow + kb + 16 + lhalf * 8);
    acc = __builtin_amdgcn_wmma_f32_16x16x32_f16(false, a, false, bm,
                                                 (short)0, acc, false, false);
  }

  // --- writeout: even->dy, odd->dx, 18+k -> fast sigmoid -> mask ---
  const int n  = nbase + lmod;
  const int py = ty0 + (n >> 3), px = tx0 + (n & 7);
  const size_t obase = (size_t)b * 9 * HWSZ + (size_t)py * WW_ + px;
#pragma unroll
  for (int r = 0; r < 8; ++r) {
    const int m = mbase + lhalf * 8 + r;
    if (m < 27) {
      const float v = acc[r] + b_off[m];
      if (m < 18) {
        ((m & 1) ? dx : dy)[obase + (size_t)(m >> 1) * HWSZ] = v;
      } else {
        msk[obase + (size_t)(m - 18) * HWSZ] =
            __builtin_amdgcn_rcpf(1.0f + __expf(-v));
      }
    }
  }
}

// ------------- kernel 2: modulated bilinear gather + main WMMA GEMM ---------
extern "C" __global__ void __launch_bounds__(256)
dcn_main_kernel(const _Float16* __restrict__ x16,
                const float* __restrict__ dy, const float* __restrict__ dx,
                const float* __restrict__ msk,
                const _Float16* __restrict__ Adcn, float* __restrict__ out) {
  extern __shared__ char smem[];
  v4i* sI      = (v4i*)smem;                       // [576] corner element offsets
  v4f* sW      = (v4f*)(smem + 9216);              // [576] modulated bilinear weights
  _Float16* sB = (_Float16*)(smem + 18432);        // [64 px][KPITCH] f16 patches

  const int tile = blockIdx.x;
  const int b    = blockIdx.y;
  const int ty0  = (tile >> 4) * 8, tx0 = (tile & 15) * 8;
  const int t    = threadIdx.x;

  // --- phase 0: one thread per pixel precomputes corners + weights (OOB folded) ---
  if (t < 64) {
    const int p  = t;
    const int py = ty0 + (p >> 3), px = tx0 + (p & 7);
    const size_t obase = (size_t)b * 9 * HWSZ + (size_t)py * WW_ + px;
#pragma unroll
    for (int kk = 0; kk < 9; ++kk) {
      const float offy = dy[obase + (size_t)kk * HWSZ];
      const float offx = dx[obase + (size_t)kk * HWSZ];
      const float mk   = msk[obase + (size_t)kk * HWSZ];
      const float fy = (float)(py - 1 + kk / 3) + offy;
      const float fx = (float)(px - 1 + kk % 3) + offx;
      const float y0f = floorf(fy), x0f = floorf(fx);
      const int   y0 = (int)y0f,    x0 = (int)x0f;
      const float wy1 = fy - y0f, wx1 = fx - x0f;
      const float wy0 = 1.0f - wy1, wx0 = 1.0f - wx1;
      const bool vy0 = (y0 >= 0) && (y0 < HH);
      const bool vy1 = (y0 + 1 >= 0) && (y0 + 1 < HH);
      const bool vx0 = (x0 >= 0) && (x0 < WW_);
      const bool vx1 = (x0 + 1 >= 0) && (x0 + 1 < WW_);
      const int cy0 = y0 < 0 ? 0 : (y0 > HH - 1 ? HH - 1 : y0);
      const int cy1 = (y0 + 1) < 0 ? 0 : ((y0 + 1) > HH - 1 ? HH - 1 : y0 + 1);
      const int cx0 = x0 < 0 ? 0 : (x0 > WW_ - 1 ? WW_ - 1 : x0);
      const int cx1 = (x0 + 1) < 0 ? 0 : ((x0 + 1) > WW_ - 1 ? WW_ - 1 : x0 + 1);
      v4i idx;
      idx[0] = (cy0 * WW_ + cx0) * CIN;
      idx[1] = (cy0 * WW_ + cx1) * CIN;
      idx[2] = (cy1 * WW_ + cx0) * CIN;
      idx[3] = (cy1 * WW_ + cx1) * CIN;
      v4f w;
      w[0] = (vy0 && vx0) ? mk * wy0 * wx0 : 0.0f;
      w[1] = (vy0 && vx1) ? mk * wy0 * wx1 : 0.0f;
      w[2] = (vy1 && vx0) ? mk * wy1 * wx0 : 0.0f;
      w[3] = (vy1 && vx1) ? mk * wy1 * wx1 : 0.0f;
      sI[kk * 64 + p] = idx;
      sW[kk * 64 + p] = w;
    }
  }
  __syncthreads();

  { // --- phase 1: packed-f16 modulated gather from channel-last x16 ---
    const int p  = t & 63, cg = t >> 6;
    const _Float16* xb = x16 + (size_t)b * HWSZ * CIN + cg * 16;
#pragma unroll
    for (int kk = 0; kk < 9; ++kk) {
      const v4i idx = sI[kk * 64 + p];     // broadcast ds_load_b128
      const v4f w   = sW[kk * 64 + p];
      const v8h w0 = splat8(w[0]), w1 = splat8(w[1]);
      const v8h w2 = splat8(w[2]), w3 = splat8(w[3]);
      const _Float16* p00 = xb + idx[0];
      const _Float16* p01 = xb + idx[1];
      const _Float16* p10 = xb + idx[2];
      const _Float16* p11 = xb + idx[3];
      const v8h a00 = *(const v8h*)p00, b00 = *(const v8h*)(p00 + 8);
      const v8h a01 = *(const v8h*)p01, b01 = *(const v8h*)(p01 + 8);
      const v8h a10 = *(const v8h*)p10, b10 = *(const v8h*)(p10 + 8);
      const v8h a11 = *(const v8h*)p11, b11 = *(const v8h*)(p11 + 8);
      const v8h ra = a00 * w0 + a01 * w1 + a10 * w2 + a11 * w3;  // v_pk_fma_f16
      const v8h rb = b00 * w0 + b01 * w1 + b10 * w2 + b11 * w3;
      _Float16* d = sB + p * KPITCH + kk * 64 + cg * 16;
      *(v8h*)d = ra;
      *(v8h*)(d + 8) = rb;
    }
  }
  __syncthreads();

  // --- GEMM: M=64 x N=64 x K=576; 8 waves, one M-block x two N-blocks each ---
  const int wave  = t >> 5, lane = t & 31;
  const int lhalf = lane >> 4, lmod = lane & 15;
  const int mbase = (wave & 3) * 16;
  const int nb0   = (wave >> 2) * 2;
  v8f acc0 = {}, acc1 = {};
  const _Float16* arow  = Adcn + (size_t)(mbase + lmod) * KDIM;
  const _Float16* brow0 = sB + (size_t)(nb0 * 16 + lmod) * KPITCH;
  const _Float16* brow1 = brow0 + 16 * KPITCH;
#pragma unroll 2
  for (int kb = 0; kb < KDIM; kb += 32) {
    const v16h a  = make_frag(arow  + kb + lhalf * 8, arow  + kb + 16 + lhalf * 8);
    const v16h b0 = make_frag(brow0 + kb + lhalf * 8, brow0 + kb + 16 + lhalf * 8);
    const v16h b1 = make_frag(brow1 + kb + lhalf * 8, brow1 + kb + 16 + lhalf * 8);
    acc0 = __builtin_amdgcn_wmma_f32_16x16x32_f16(false, a, false, b0,
                                                  (short)0, acc0, false, false);
    acc1 = __builtin_amdgcn_wmma_f32_16x16x32_f16(false, a, false, b1,
                                                  (short)0, acc1, false, false);
  }

  // --- writeout: C/D layout -> out[b][m][py][px] ---
#pragma unroll
  for (int blkn = 0; blkn < 2; ++blkn) {
    const v8f acc = blkn ? acc1 : acc0;
    const int n  = (nb0 + blkn) * 16 + lmod;
    const int py = ty0 + (n >> 3), px = tx0 + (n & 7);
#pragma unroll
    for (int r = 0; r < 8; ++r) {
      const int m = mbase + lhalf * 8 + r;
      out[((size_t)b * COUT + m) * HWSZ + (size_t)py * WW_ + px] = acc[r];
    }
  }
}

// ------------------------------- launcher -----------------------------------
extern "C" void kernel_launch(void* const* d_in, const int* in_sizes, int n_in,
                              void* d_out, int out_size, void* d_ws, size_t ws_size,
                              hipStream_t stream) {
  const float* x        = (const float*)d_in[0];   // [8,64,128,128]
  const float* w_offset = (const float*)d_in[1];   // [27,64,3,3]
  const float* b_offset = (const float*)d_in[2];   // [27]
  const float* w_dcn    = (const float*)d_in[3];   // [64,64,3,3]
  float* out = (float*)d_out;                      // [8,64,128,128]

  // workspace layout (all regions fully rewritten every call)
  char* ws = (char*)d_ws;
  _Float16* Aoff = (_Float16*)ws;                          //     36,864 B
  _Float16* Adcn = (_Float16*)(ws + 36864);                //     73,728 B
  _Float16* x16  = (_Float16*)(ws + 110592);               // 16,777,216 B
  float* dyp = (float*)(ws + 110592 + (size_t)16777216);   //  4,718,592 B
  float* dxp = dyp + (size_t)8 * 9 * HWSZ;
  float* mkp = dxp + (size_t)8 * 9 * HWSZ;

  const size_t sh1 = (size_t)64 * KPITCH * sizeof(_Float16);        // 74,752 B
  const size_t sh2 = 18432 + sh1;                                   // 93,184 B (<320KB/WGP)

  xcvt_kernel<<<(8 * HWSZ) / 256, 256, 0, stream>>>(x, x16);
  wconvert_kernel<<<(64 * KDIM + 255) / 256, 256, 0, stream>>>(w_offset, w_dcn, Aoff, Adcn);
  offset_conv_kernel<<<dim3(256, 8), 256, sh1, stream>>>(x16, b_offset, Aoff, dyp, dxp, mkp);
  dcn_main_kernel<<<dim3(256, 8), 256, sh2, stream>>>(x16, dyp, dxp, mkp, Adcn, out);
}
